// CausalSelfAttention_64707977281590
// MI455X (gfx1250) — matrix operand order repaired and checked by reading, not compiled, not run on hardware
//
#include <hip/hip_runtime.h>

// CDNA5 (gfx1250) wave32 WMMA causal self-attention.
// B=2, T=2048, C=1024, H=16, D=64.  M = B*T = 4096.

typedef __attribute__((ext_vector_type(16))) _Float16     v16h;
typedef __attribute__((ext_vector_type(8)))  _Float16     v8h;
typedef __attribute__((ext_vector_type(4)))  _Float16     v4h;
typedef __attribute__((ext_vector_type(8)))  float        v8f;
typedef __attribute__((ext_vector_type(4)))  unsigned int v4u;
typedef __attribute__((ext_vector_type(8)))  int          v8i;
typedef __attribute__((ext_vector_type(4)))  int          v4i;

#define NB 2
#define NT 2048
#define NC 1024
#define NH 16
#define ND 64
#define NM (NB * NT)   // 4096

static __device__ __forceinline__ v8f wmma_f16(v16h a, v16h b, v8f c) {
    return __builtin_amdgcn_wmma_f32_16x16x32_f16(false, a, false, b, (short)0, c, false, false);
}

static __device__ __forceinline__ v16h ldsv16(const _Float16* p) {
    v8h lo = *(const v8h*)p;
    v8h hi = *(const v8h*)(p + 8);
    return __builtin_shufflevector(lo, hi, 0, 1, 2, 3, 4, 5, 6, 7,
                                           8, 9, 10, 11, 12, 13, 14, 15);
}

// A fragment: 16x32 f16 tile, element (r,c) at p[r*ld+c].
static __device__ __forceinline__ v16h a_frag(const _Float16* p, int ld, int lane) {
    const int r  = lane & 15;
    const int kh = (lane >> 4) * 16;
    const _Float16* base = p + r * ld + kh;
    v8h lo = *(const v8h*)(base);        // K = kh .. kh+7
    v8h hi = *(const v8h*)(base + 8);    // K = kh+8 .. kh+15
    return __builtin_shufflevector(lo, hi, 0, 1, 2, 3, 4, 5, 6, 7,
                                           8, 9, 10, 11, 12, 13, 14, 15);
}

// B fragment: 32x16 f16 tile stored K-major: element (k,n) at p[n*ld + k].
static __device__ __forceinline__ v16h b_frag(const _Float16* p, int ld, int lane) {
    return ldsv16(p + (lane & 15) * ld + (lane >> 4) * 16);
}

// C/D layout: acc[v] holds element (M = 8*(lane>=16) + v, N = lane&15).

// ---------------------------------------------------------------------------
// Tensor Data Mover: 2D f16 tile (tile_d0 x tile_d1) global -> LDS.
// ---------------------------------------------------------------------------
#if defined(__has_builtin)
#  if __has_builtin(__builtin_amdgcn_tensor_load_to_lds)
#    define HAVE_TDM 1
#  endif
#  if __has_builtin(__builtin_amdgcn_s_wait_tensorcnt)
#    define HAVE_TENSORCNT 1
#  endif
#endif

#ifdef HAVE_TDM
static __device__ __forceinline__ void tdm_load_2d_f16(
    unsigned lds_off, const _Float16* gptr,
    unsigned tile_d0, unsigned tile_d1, unsigned stride0)
{
    unsigned long long ga = (unsigned long long)(const void*)gptr;
    v4u g0;
    g0[0] = 1u;                                            // count=1 (valid user D#)
    g0[1] = lds_off;                                       // lds_addr (bytes)
    g0[2] = (unsigned)ga;                                  // global_addr[31:0]
    g0[3] = (unsigned)((ga >> 32) & 0x1FFFFFFull) | (2u << 30); // addr[56:32] | type=2
    v8i g1;
    g1[0] = (int)(1u << 16);                               // data_size=1 -> 2 bytes
    g1[1] = (int)((tile_d0 & 0xFFFFu) << 16);              // tensor_dim0[15:0]
    g1[2] = (int)(tile_d1 << 16);                          // tensor_dim0 hi | tensor_dim1 lo
    g1[3] = (int)(tile_d0 << 16);                          // tensor_dim1 hi | tile_dim0
    g1[4] = (int)(tile_d1 & 0xFFFFu);                      // tile_dim1 | tile_dim2=0
    g1[5] = (int)stride0;                                  // tensor_dim0_stride[31:0]
    g1[6] = 0;
    g1[7] = 0;
    v4i z = {0, 0, 0, 0};
#if __clang_major__ <= 22
    __builtin_amdgcn_tensor_load_to_lds(g0, g1, z, z, 0);
#else
    v8i z8 = {0, 0, 0, 0, 0, 0, 0, 0};
    __builtin_amdgcn_tensor_load_to_lds(g0, g1, z, z, z8, 0);
#endif
}
#endif  // HAVE_TDM

// ---------------------------------------------------------------------------
// Kernel 1: qkv = x @ qkv_w + qkv_b ; q,k -> [B,H,T,D], v -> [B,H,D,T] (f16).
// Grid: (3072/64, 4096/64); block 128. Double-buffered LDS, register-staged.
// ---------------------------------------------------------------------------
__global__ __launch_bounds__(128) void qkv_gemm_kernel(
    const float* __restrict__ x, const float* __restrict__ w,
    const float* __restrict__ bias,
    _Float16* __restrict__ oq, _Float16* __restrict__ ok, _Float16* __restrict__ ov)
{
    __shared__ __align__(16) _Float16 sA[2][64 * 32];   // row-major
    __shared__ __align__(16) _Float16 sBt[2][64 * 32];  // transposed: [n][k]
    const int tid  = threadIdx.x;
    const int lane = tid & 31;
    const int wv   = tid >> 5;
    const int m0   = blockIdx.y * 64;
    const int n0   = blockIdx.x * 64;

    float4 fa[4], fb[4];
    auto g_load = [&](int kk) {
#pragma unroll
        for (int u = 0; u < 4; ++u) {
            int i = tid + u * 128;
            int ra = i >> 3, ca = (i & 7) * 4;
            fa[u] = *(const float4*)&x[(size_t)(m0 + ra) * NC + kk + ca];
            int rb = i >> 4, cb = (i & 15) * 4;
            fb[u] = *(const float4*)&w[(size_t)(kk + rb) * (3 * NC) + n0 + cb];
        }
    };
    auto s_store = [&](int buf) {
        _Float16* dA = sA[buf];
        _Float16* dB = sBt[buf];
#pragma unroll
        for (int u = 0; u < 4; ++u) {
            int i = tid + u * 128;
            int ra = i >> 3, ca = (i & 7) * 4;
            v4h h = { (_Float16)fa[u].x, (_Float16)fa[u].y,
                      (_Float16)fa[u].z, (_Float16)fa[u].w };
            *(v4h*)&dA[ra * 32 + ca] = h;
            int rb = i >> 4, cb = (i & 15) * 4;
            dB[(cb + 0) * 32 + rb] = (_Float16)fb[u].x;
            dB[(cb + 1) * 32 + rb] = (_Float16)fb[u].y;
            dB[(cb + 2) * 32 + rb] = (_Float16)fb[u].z;
            dB[(cb + 3) * 32 + rb] = (_Float16)fb[u].w;
        }
    };

    v8f acc[4] = {};
    g_load(0);
    s_store(0);
    int buf = 0;
    for (int kk = 0; kk < NC; kk += 32, buf ^= 1) {
        if (kk + 32 < NC) g_load(kk + 32);     // next tile's loads in flight
        __syncthreads();                        // current tile visible
        v16h a = a_frag(sA[buf] + wv * 16 * 32, 32, lane);
#pragma unroll
        for (int t = 0; t < 4; ++t) {
            v16h b = b_frag(sBt[buf] + (t * 16) * 32, 32, lane);
            acc[t] = wmma_f16(a, b, acc[t]);
        }
        if (kk + 32 < NC) s_store(buf ^ 1);     // fill other buffer
    }
#pragma unroll
    for (int t = 0; t < 4; ++t) {
#pragma unroll
        for (int e = 0; e < 8; ++e) {
            int mg = m0 + wv * 16 + 8 * (lane >> 4) + e;
            int ng = n0 + t * 16 + (lane & 15);
            float val   = acc[t][e] + bias[ng];
            int   which = ng >> 10;          // 0=q 1=k 2=v
            int   cc    = ng & (NC - 1);
            int   hh    = cc >> 6, dd = cc & 63;
            int   bb    = mg >> 11, tt = mg & (NT - 1);
            size_t head = (size_t)bb * NH + hh;
            if (which == 2)
                ov[(head * ND + dd) * NT + tt] = (_Float16)val;   // V transposed
            else {
                _Float16* dst = (which == 0) ? oq : ok;
                dst[(head * NT + tt) * ND + dd] = (_Float16)val;
            }
        }
    }
}

// ---------------------------------------------------------------------------
// Kernel 2: causal flash attention per (b,h). Grid: (T/64, B*H); block 128.
// K/V tiles: TDM, double-buffered and software-pipelined on TENSORcnt.
// ---------------------------------------------------------------------------
__global__ __launch_bounds__(128) void attn_kernel(
    const _Float16* __restrict__ Q, const _Float16* __restrict__ K,
    const _Float16* __restrict__ Vt, _Float16* __restrict__ att)
{
    __shared__ __align__(16) _Float16 sK[2][64 * 64];    // K rows, row-major
    __shared__ __align__(16) _Float16 sVt[2][64 * 64];   // V^T: [d][kidx]
    __shared__ __align__(16) _Float16 sP[4 * 16 * 64];

    const int tid  = threadIdx.x;
    const int lane = tid & 31;
    const int wv   = tid >> 5;
    const int qb   = blockIdx.x;           // query block 0..31
    const int bh   = blockIdx.y;           // 0..31
    const int bb   = bh >> 4, hh = bh & 15;

    const size_t head_off = (size_t)bh * NT * ND;
    const _Float16* Qh  = Q  + head_off;
    const _Float16* Kh  = K  + head_off;
    const _Float16* Vth = Vt + head_off;   // [D][T]

    const int q0 = qb * 64 + wv * 16;

    v16h qf[2];
#pragma unroll
    for (int s = 0; s < 2; ++s)
        qf[s] = a_frag(Qh + (size_t)q0 * ND + s * 32, ND, lane);

    v8f   o[4] = {};
    float rmax[8], rsum[8];
#pragma unroll
    for (int e = 0; e < 8; ++e) { rmax[e] = -1e30f; rsum[e] = 0.0f; }

#ifdef HAVE_TDM
    if (wv == 0) {   // prologue: block 0 tiles into buffer 0
        tdm_load_2d_f16((unsigned)(unsigned long long)(const void*)&sK[0][0],
                        Kh, 64, 64, ND);
        tdm_load_2d_f16((unsigned)(unsigned long long)(const void*)&sVt[0][0],
                        Vth, 64, 64, NT);
    }
#endif

    for (int j = 0; j <= qb; ++j) {
        const int kb  = j * 64;
        const int buf = j & 1;
        __syncthreads();   // all readers of the buffer being refilled are done
#ifdef HAVE_TDM
        if (wv == 0) {
            if (j < qb) {  // issue next block into other buffer, keep in flight
                tdm_load_2d_f16((unsigned)(unsigned long long)(const void*)&sK[buf ^ 1][0],
                                Kh + (size_t)(kb + 64) * ND, 64, 64, ND);
                tdm_load_2d_f16((unsigned)(unsigned long long)(const void*)&sVt[buf ^ 1][0],
                                Vth + kb + 64, 64, 64, NT);
#ifdef HAVE_TENSORCNT
                __builtin_amdgcn_s_wait_tensorcnt(2);   // block j's pair done
#else
                asm volatile("s_wait_tensorcnt 0x2" ::: "memory");
#endif
            } else {
#ifdef HAVE_TENSORCNT
                __builtin_amdgcn_s_wait_tensorcnt(0);
#else
                asm volatile("s_wait_tensorcnt 0x0" ::: "memory");
#endif
            }
        }
#else
        for (int i = tid; i < 64 * 64 / 8; i += 128) {
            int r = i >> 3, c8 = (i & 7) * 8;
            *(v8h*)&sK[buf][r * 64 + c8]  = *(const v8h*)&Kh[(size_t)(kb + r) * ND + c8];
            *(v8h*)&sVt[buf][r * 64 + c8] = *(const v8h*)&Vth[(size_t)r * NT + kb + c8];
        }
#endif
        __syncthreads();

        // S = Q K^T  (16 x 64 strip, fp32 accum)
        v8f s_acc[4] = {};
#pragma unroll
        for (int t = 0; t < 4; ++t) {
#pragma unroll
            for (int s = 0; s < 2; ++s) {
                v16h b = b_frag(sK[buf] + (t * 16) * 64 + s * 32, 64, lane);
                s_acc[t] = wmma_f16(qf[s], b, s_acc[t]);
            }
        }

        // scale 1/sqrt(D); causal mask on diagonal block
#pragma unroll
        for (int t = 0; t < 4; ++t) {
            int kg = kb + t * 16 + (lane & 15);
#pragma unroll
            for (int e = 0; e < 8; ++e) {
                float sv = s_acc[t][e] * 0.125f;
                int   qg = q0 + 8 * (lane >> 4) + e;
                s_acc[t][e] = (j == qb && kg > qg) ? -1e30f : sv;
            }
        }

        // online softmax; P (f16) into per-wave LDS strip
        _Float16* pw = sP + wv * (16 * 64);
#pragma unroll
        for (int e = 0; e < 8; ++e) {
            float m = fmaxf(fmaxf(s_acc[0][e], s_acc[1][e]),
                            fmaxf(s_acc[2][e], s_acc[3][e]));
#pragma unroll
            for (int sx = 1; sx < 16; sx <<= 1)
                m = fmaxf(m, __shfl_xor(m, sx, 32));
            float nmax = fmaxf(rmax[e], m);
            float corr = __expf(rmax[e] - nmax);
            rmax[e] = nmax;

            int   prow = 8 * (lane >> 4) + e;
            float sum  = 0.0f;
#pragma unroll
            for (int t = 0; t < 4; ++t) {
                float p = __expf(s_acc[t][e] - nmax);
                sum += p;
                pw[prow * 64 + t * 16 + (lane & 15)] = (_Float16)p;
            }
#pragma unroll
            for (int sx = 1; sx < 16; sx <<= 1)
                sum += __shfl_xor(sum, sx, 32);
            rsum[e] = rsum[e] * corr + sum;
#pragma unroll
            for (int t = 0; t < 4; ++t)
                o[t][e] *= corr;
        }
        __syncthreads();   // cross-lane P visibility (uniform bound -> legal)

        // O += P V
#pragma unroll
        for (int s = 0; s < 2; ++s) {
            v16h pf = a_frag(pw + s * 32, 64, lane);
#pragma unroll
            for (int t = 0; t < 4; ++t) {
                v16h b = b_frag(sVt[buf] + (t * 16) * 64 + s * 32, 64, lane);
                o[t] = wmma_f16(pf, b, o[t]);
            }
        }
    }

    // normalize; write att [B,T,H,D] == [B*T, C] with c = h*64+d
#pragma unroll
    for (int t = 0; t < 4; ++t) {
#pragma unroll
        for (int e = 0; e < 8; ++e) {
            int   qg  = q0 + 8 * (lane >> 4) + e;
            int   dd  = t * 16 + (lane & 15);
            float val = o[t][e] / rsum[e];
            att[((size_t)(bb * NT + qg) * NH + hh) * ND + dd] = (_Float16)val;
        }
    }
}

// ---------------------------------------------------------------------------
// Kernel 3: out = att @ proj_w + proj_b (fp32). Grid: (1024/64, 4096/64).
// ---------------------------------------------------------------------------
__global__ __launch_bounds__(128) void proj_gemm_kernel(
    const _Float16* __restrict__ A, const float* __restrict__ w,
    const float* __restrict__ bias, float* __restrict__ out)
{
    __shared__ __align__(16) _Float16 sA[2][64 * 32];
    __shared__ __align__(16) _Float16 sBt[2][64 * 32];
    const int tid  = threadIdx.x;
    const int lane = tid & 31;
    const int wv   = tid >> 5;
    const int m0   = blockIdx.y * 64;
    const int n0   = blockIdx.x * 64;

    v8h    ha[2];
    float4 fb[4];
    auto g_load = [&](int kk) {
#pragma unroll
        for (int u = 0; u < 2; ++u) {
            int i = tid + u * 128;
            int ra = i >> 2, ca = (i & 3) * 8;
            ha[u] = *(const v8h*)&A[(size_t)(m0 + ra) * NC + kk + ca];
        }
#pragma unroll
        for (int u = 0; u < 4; ++u) {
            int i = tid + u * 128;
            int rb = i >> 4, cb = (i & 15) * 4;
            fb[u] = *(const float4*)&w[(size_t)(kk + rb) * NC + n0 + cb];
        }
    };
    auto s_store = [&](int buf) {
        _Float16* dA = sA[buf];
        _Float16* dB = sBt[buf];
#pragma unroll
        for (int u = 0; u < 2; ++u) {
            int i = tid + u * 128;
            int ra = i >> 2, ca = (i & 3) * 8;
            *(v8h*)&dA[ra * 32 + ca] = ha[u];
        }
#pragma unroll
        for (int u = 0; u < 4; ++u) {
            int i = tid + u * 128;
            int rb = i >> 4, cb = (i & 15) * 4;
            dB[(cb + 0) * 32 + rb] = (_Float16)fb[u].x;
            dB[(cb + 1) * 32 + rb] = (_Float16)fb[u].y;
            dB[(cb + 2) * 32 + rb] = (_Float16)fb[u].z;
            dB[(cb + 3) * 32 + rb] = (_Float16)fb[u].w;
        }
    };

    v8f acc[4] = {};
    g_load(0);
    s_store(0);
    int buf = 0;
    for (int kk = 0; kk < NC; kk += 32, buf ^= 1) {
        if (kk + 32 < NC) g_load(kk + 32);
        __syncthreads();
        v16h a = a_frag(sA[buf] + wv * 16 * 32, 32, lane);
#pragma unroll
        for (int t = 0; t < 4; ++t) {
            v16h b = b_frag(sBt[buf] + (t * 16) * 32, 32, lane);
            acc[t] = wmma_f16(a, b, acc[t]);
        }
        if (kk + 32 < NC) s_store(buf ^ 1);
    }
#pragma unroll
    for (int t = 0; t < 4; ++t) {
#pragma unroll
        for (int e = 0; e < 8; ++e) {
            int mg = m0 + wv * 16 + 8 * (lane >> 4) + e;
            int ng = n0 + t * 16 + (lane & 15);
            out[(size_t)mg * NC + ng] = acc[t][e] + bias[ng];
        }
    }
}

// ---------------------------------------------------------------------------
extern "C" void kernel_launch(void* const* d_in, const int* in_sizes, int n_in,
                              void* d_out, int out_size, void* d_ws, size_t ws_size,
                              hipStream_t stream)
{
    (void)in_sizes; (void)n_in; (void)out_size; (void)ws_size;
    const float* x      = (const float*)d_in[0];
    const float* qkv_w  = (const float*)d_in[1];
    const float* qkv_b  = (const float*)d_in[2];
    const float* proj_w = (const float*)d_in[3];
    const float* proj_b = (const float*)d_in[4];
    float* out = (float*)d_out;

    char* ws = (char*)d_ws;
    const size_t buf = (size_t)NB * NH * NT * ND * sizeof(_Float16);  // 8 MB
    _Float16* wq  = (_Float16*)(ws + 0 * buf);
    _Float16* wk  = (_Float16*)(ws + 1 * buf);
    _Float16* wvt = (_Float16*)(ws + 2 * buf);   // V transposed [B,H,D,T]
    _Float16* wa  = (_Float16*)(ws + 3 * buf);

    qkv_gemm_kernel<<<dim3(3 * NC / 64, NM / 64), 128, 0, stream>>>(x, qkv_w, qkv_b, wq, wk, wvt);
    attn_kernel<<<dim3(NT / 64, NB * NH), 128, 0, stream>>>(wq, wk, wvt, wa);
    proj_gemm_kernel<<<dim3(NC / 64, NM / 64), 128, 0, stream>>>(wa, proj_w, proj_b, out);
}
